// StructuralClusterPredictor_5093831213709
// MI455X (gfx1250) — compile-verified, matrix operand order checked by reading, not run on hardware
//
#include <hip/hip_runtime.h>
#include <hip/hip_bf16.h>
#include <math.h>

typedef __attribute__((ext_vector_type(16))) _Float16 v16h;
typedef __attribute__((ext_vector_type(8)))  float    v8f;
typedef __attribute__((ext_vector_type(4)))  float    v4f;

#define BB 32
#define NN 4096
#define DD 512
#define IN_DIM 1541            // 3*512 + 5
#define KP 1568                // padded to 49*32 for WMMA K-chunks
#define HID 64
#define NS 16                  // N-dimension split for the big reduction
#define NCHUNK (NN / NS)       // 256
#define D4 (DD / 4)            // 128 float4 columns
#define NEG_SENT -100000000.0f

// ---------------------------------------------------------------------------
// Kernel 1: per-batch valid count + scalar features + zero-pad tail of g
// ---------------------------------------------------------------------------
__global__ void count_sf_kernel(const float* __restrict__ mask,
                                float* __restrict__ cnt,
                                float* __restrict__ g) {
  int b = threadIdx.x;
  if (b >= BB) return;
  const float* mrow = mask + (size_t)b * NN;
  float c = 0.f;
  #pragma unroll 8
  for (int n = 0; n < NN; ++n) c += (mrow[n] > NEG_SENT) ? 1.f : 0.f;
  cnt[b] = c;
  float* gr = g + (size_t)b * KP;
  gr[3 * DD + 0] = logf(c + 1.f) * 0.2f;
  gr[3 * DD + 1] = 0.1f;
  gr[3 * DD + 2] = 0.2f;
  gr[3 * DD + 3] = 4.0f;
  gr[3 * DD + 4] = 0.1f;
  for (int k = IN_DIM; k < KP; ++k) gr[k] = 0.f;
}

// ---------------------------------------------------------------------------
// Kernel 2: transpose + f16-convert weights.  W1T[n][k] (64 x KP, zero pad),
// W2T[n][k] (32 x 64).
// ---------------------------------------------------------------------------
__global__ void prep_weights_kernel(const float* __restrict__ W1,
                                    const float* __restrict__ W2,
                                    _Float16* __restrict__ w1t,
                                    _Float16* __restrict__ w2t) {
  int t = blockIdx.x * blockDim.x + threadIdx.x;
  const int n1 = HID * KP;
  if (t < n1) {
    int n = t / KP, k = t % KP;
    w1t[t] = (k < IN_DIM) ? (_Float16)W1[(size_t)k * HID + n] : (_Float16)0.0f;
  } else {
    int u = t - n1;
    if (u < 32 * 64) {
      int n = u / 64, k = u % 64;
      w2t[u] = (_Float16)W2[(size_t)k * 32 + n];
    }
  }
}

// ---------------------------------------------------------------------------
// Kernel 3: masked partial reduction over an N-chunk, b128 non-temporal
// streaming loads.  grid = (NS, BB), block = 128 threads; each thread owns
// 4 consecutive d columns (one float4 per row).  Mask validity staged in LDS.
// ---------------------------------------------------------------------------
__global__ void __launch_bounds__(128)
partial_reduce_kernel(const float* __restrict__ x,
                      const float* __restrict__ mask,
                      float* __restrict__ psum,
                      float* __restrict__ psq,
                      float* __restrict__ pmax) {
  __shared__ float lm[NCHUNK];
  const int s = blockIdx.x;
  const int b = blockIdx.y;
  const int tid = threadIdx.x;            // 0..127 -> float4 column
  const int n0 = s * NCHUNK;

  for (int i = tid; i < NCHUNK; i += blockDim.x)
    lm[i] = (mask[(size_t)b * NN + n0 + i] > NEG_SENT) ? 1.f : 0.f;
  __syncthreads();

  const v4f* xp = (const v4f*)(x + ((size_t)b * NN + n0) * DD) + tid;
  v4f sum = {}, sq = {};
  v4f mx = {-INFINITY, -INFINITY, -INFINITY, -INFINITY};
  #pragma unroll 8
  for (int i = 0; i < NCHUNK; ++i) {
    v4f v = __builtin_nontemporal_load(xp + (size_t)i * D4);
    float m = lm[i];
    sum += m * v;
    sq  += m * (v * v);
    #pragma unroll
    for (int c = 0; c < 4; ++c)
      mx[c] = (m > 0.f) ? fmaxf(mx[c], v[c]) : mx[c];
  }
  size_t o = ((size_t)(b * NS + s)) * D4 + tid;
  ((v4f*)psum)[o] = sum;
  ((v4f*)psq)[o]  = sq;
  ((v4f*)pmax)[o] = mx;
}

// ---------------------------------------------------------------------------
// Kernel 4: combine NS partials -> mean / max / std, write into g (vectorized).
// ---------------------------------------------------------------------------
__global__ void combine_kernel(const float* __restrict__ psum,
                               const float* __restrict__ psq,
                               const float* __restrict__ pmax,
                               const float* __restrict__ cnt,
                               float* __restrict__ g) {
  int t = blockIdx.x * blockDim.x + threadIdx.x;
  if (t >= BB * D4) return;
  int b = t / D4, d4 = t % D4;
  v4f sum = {}, sq = {};
  v4f mx = {-INFINITY, -INFINITY, -INFINITY, -INFINITY};
  #pragma unroll
  for (int s = 0; s < NS; ++s) {
    size_t o = ((size_t)(b * NS + s)) * D4 + d4;
    sum += ((const v4f*)psum)[o];
    sq  += ((const v4f*)psq)[o];
    v4f pm = ((const v4f*)pmax)[o];
    #pragma unroll
    for (int c = 0; c < 4; ++c) mx[c] = fmaxf(mx[c], pm[c]);
  }
  float ccnt = cnt[b];
  float cs   = fmaxf(ccnt, 1.f);
  bool  has  = ccnt > 0.f;
  v4f mean, mmax, stdv;
  #pragma unroll
  for (int c = 0; c < 4; ++c) {
    float me = has ? (sum[c] / cs) : 0.f;
    float va = fmaxf((sq[c] - ccnt * me * me) / fmaxf(ccnt - 1.f, 1.f), 0.f);
    mean[c] = me;
    mmax[c] = has ? mx[c] : 0.f;
    stdv[c] = has ? sqrtf(va) : 0.f;
  }
  float* gr = g + (size_t)b * KP + 4 * d4;     // 16-byte aligned (KP%4==0)
  *(v4f*)(gr)            = mean;
  *(v4f*)(gr + DD)       = mmax;
  *(v4f*)(gr + 2 * DD)   = stdv;
}

// ---------------------------------------------------------------------------
// Kernel 5: one-wave WMMA MLP.  M=32 (2 tiles) x N=64 (4 tiles), K=1568.
// Then 32x64 @ 64x32 (2x2 tiles), then per-lane dot for the final layer.
// Fragment layouts per CDNA5 ISA §7.12.2 (wave32).
// ---------------------------------------------------------------------------
__global__ void __launch_bounds__(32)
mlp_wmma_kernel(const float* __restrict__ g,
                const _Float16* __restrict__ w1t,
                const _Float16* __restrict__ w2t,
                const float* __restrict__ b1,
                const float* __restrict__ b2,
                const float* __restrict__ W3,
                const float* __restrict__ b3,
                float* __restrict__ out) {
  __shared__ float h1[32][HID];
  __shared__ float h2[32][32];
  __shared__ float red_s[32];
  __shared__ float red_r[32];

  const int lane = threadIdx.x;
  const int r16  = lane & 15;
  const int hi   = (lane >> 4) & 1;       // upper half-wave
  const int k0   = hi ? 8 : 0;            // start of first 8-element K-run

  v8f zero = {};
  v8f acc[2][4];
  #pragma unroll
  for (int i = 0; i < 2; ++i)
    #pragma unroll
    for (int j = 0; j < 4; ++j) acc[i][j] = zero;

  // ---- layer 1: g(32 x 1568) @ W1T^T (1568 x 64) ----
  for (int kc = 0; kc < KP; kc += 32) {
    v16h afr[2], bfr[4];
    #pragma unroll
    for (int mt = 0; mt < 2; ++mt) {
      const float* gp = g + (size_t)(mt * 16 + r16) * KP + kc + k0;
      #pragma unroll
      for (int h = 0; h < 8; ++h) {
        afr[mt][h]     = (_Float16)gp[h];
        afr[mt][h + 8] = (_Float16)gp[16 + h];
      }
    }
    #pragma unroll
    for (int nt = 0; nt < 4; ++nt) {
      const _Float16* wp = w1t + (size_t)(nt * 16 + r16) * KP + kc + k0;
      #pragma unroll
      for (int h = 0; h < 8; ++h) {
        bfr[nt][h]     = wp[h];
        bfr[nt][h + 8] = wp[16 + h];
      }
    }
    #pragma unroll
    for (int mt = 0; mt < 2; ++mt)
      #pragma unroll
      for (int nt = 0; nt < 4; ++nt)
        acc[mt][nt] = __builtin_amdgcn_wmma_f32_16x16x32_f16(
            false, afr[mt], false, bfr[nt], (short)0, acc[mt][nt], false, false);
  }

  // bias + relu -> h1 (C/D layout: row m = vgpr + (lane<16?0:8))
  #pragma unroll
  for (int mt = 0; mt < 2; ++mt)
    #pragma unroll
    for (int nt = 0; nt < 4; ++nt) {
      int n = nt * 16 + r16;
      #pragma unroll
      for (int r = 0; r < 8; ++r) {
        int m = mt * 16 + r + hi * 8;
        h1[m][n] = fmaxf(acc[mt][nt][r] + b1[n], 0.f);
      }
    }
  __syncthreads();

  // ---- layer 2: h1(32 x 64) @ W2T^T (64 x 32) ----
  v8f acc2[2][2];
  #pragma unroll
  for (int i = 0; i < 2; ++i)
    #pragma unroll
    for (int j = 0; j < 2; ++j) acc2[i][j] = zero;

  #pragma unroll
  for (int kc = 0; kc < HID; kc += 32) {
    v16h afr[2], bfr[2];
    #pragma unroll
    for (int mt = 0; mt < 2; ++mt) {
      const float* hp = &h1[mt * 16 + r16][kc + k0];
      #pragma unroll
      for (int h = 0; h < 8; ++h) {
        afr[mt][h]     = (_Float16)hp[h];
        afr[mt][h + 8] = (_Float16)hp[16 + h];
      }
    }
    #pragma unroll
    for (int nt = 0; nt < 2; ++nt) {
      const _Float16* wp = w2t + (size_t)(nt * 16 + r16) * HID + kc + k0;
      #pragma unroll
      for (int h = 0; h < 8; ++h) {
        bfr[nt][h]     = wp[h];
        bfr[nt][h + 8] = wp[16 + h];
      }
    }
    #pragma unroll
    for (int mt = 0; mt < 2; ++mt)
      #pragma unroll
      for (int nt = 0; nt < 2; ++nt)
        acc2[mt][nt] = __builtin_amdgcn_wmma_f32_16x16x32_f16(
            false, afr[mt], false, bfr[nt], (short)0, acc2[mt][nt], false, false);
  }

  #pragma unroll
  for (int mt = 0; mt < 2; ++mt)
    #pragma unroll
    for (int nt = 0; nt < 2; ++nt) {
      int n = nt * 16 + r16;
      #pragma unroll
      for (int r = 0; r < 8; ++r) {
        int m = mt * 16 + r + hi * 8;
        h2[m][n] = fmaxf(acc2[mt][nt][r] + b2[n], 0.f);
      }
    }
  __syncthreads();

  // ---- layer 3 + sigmoid + post-processing ----
  float s = b3[0];
  #pragma unroll 8
  for (int j = 0; j < 32; ++j) s += h2[lane][j] * W3[j];
  float score = 1.f / (1.f + expf(-s));
  red_s[lane] = score;
  red_r[lane] = rintf(3.f + score * 47.f);   // round per-element, then mean
  __syncthreads();
  if (lane == 0) {
    float ssum = 0.f, rsum = 0.f;
    for (int j = 0; j < 32; ++j) { ssum += red_s[j]; rsum += red_r[j]; }
    int num = (int)(rsum / 32.f);            // astype(int32) truncates
    num = num < 3 ? 3 : (num > 50 ? 50 : num);
    out[0] = (float)num;
    out[1] = ssum / 32.f;
  }
}

// ---------------------------------------------------------------------------
extern "C" void kernel_launch(void* const* d_in, const int* in_sizes, int n_in,
                              void* d_out, int out_size, void* d_ws, size_t ws_size,
                              hipStream_t stream) {
  const float* x    = (const float*)d_in[0];
  const float* mask = (const float*)d_in[1];
  const float* W1   = (const float*)d_in[2];
  const float* b1   = (const float*)d_in[3];
  const float* W2   = (const float*)d_in[4];
  const float* b2   = (const float*)d_in[5];
  const float* W3   = (const float*)d_in[6];
  const float* b3   = (const float*)d_in[7];
  float* out = (float*)d_out;

  // workspace layout (all 16B-aligned)
  float* cnt  = (float*)d_ws;                   // BB
  float* g    = cnt + BB;                       // BB*KP
  float* psum = g + (size_t)BB * KP;            // BB*NS*DD
  float* psq  = psum + (size_t)BB * NS * DD;
  float* pmax = psq  + (size_t)BB * NS * DD;
  _Float16* w1t = (_Float16*)(pmax + (size_t)BB * NS * DD);  // HID*KP
  _Float16* w2t = w1t + (size_t)HID * KP;                    // 32*64

  count_sf_kernel<<<1, 32, 0, stream>>>(mask, cnt, g);

  int prep_elems = HID * KP + 32 * 64;
  prep_weights_kernel<<<(prep_elems + 255) / 256, 256, 0, stream>>>(W1, W2, w1t, w2t);

  dim3 grid_r(NS, BB);
  partial_reduce_kernel<<<grid_r, 128, 0, stream>>>(x, mask, psum, psq, pmax);

  combine_kernel<<<(BB * D4 + 255) / 256, 256, 0, stream>>>(psum, psq, pmax, cnt, g);

  mlp_wmma_kernel<<<1, 32, 0, stream>>>(g, w1t, w2t, b1, b2, W3, b3, out);
}
